// BiBoSparseMoeBlock_75239237091863
// MI455X (gfx1250) — compile-verified, matrix operand order checked by reading, not run on hardware
//
#include <hip/hip_runtime.h>

#define TOKENS 4096
#define HIDDEN 2048
#define INTER  1024
#define NEXP   8
#define MAXM   4096      // per-expert list capacity
#define MT32   128       // MAXM / 32

typedef __attribute__((ext_vector_type(16))) __bf16 v16bf;
typedef __attribute__((ext_vector_type(8)))  float  v8f;

__device__ __forceinline__ unsigned short f2bf(float f) {
    unsigned u = __float_as_uint(f);
    unsigned r = u + 0x7fffu + ((u >> 16) & 1u);   // RNE
    return (unsigned short)(r >> 16);
}

// Load one 16x32 bf16 WMMA fragment (A or B style) from row-major,
// K-contiguous storage. `p` already includes row base + 8*(lane>>4).
// Lane h=lane>>4 supplies K = k0+8h..k0+8h+7 and k0+16+8h..k0+16+8h+7.
__device__ __forceinline__ v16bf ldfrag(const unsigned short* p, int k0) {
    union { v16bf v; uint4 q[2]; } u;
    u.q[0] = *(const uint4*)(p + k0);
    u.q[1] = *(const uint4*)(p + k0 + 16);
    return u.v;
}

__device__ __forceinline__ v8f wmma_bf16(v16bf a, v16bf b, v8f c) {
    return __builtin_amdgcn_wmma_f32_16x16x32_bf16(false, a, false, b, (short)0, c, false, false);
}

// Load a full fragment set (2 A rows-of-16 + 4 B col-tiles) at K offset k0.
__device__ __forceinline__ void ldset6(v16bf* a, v16bf* b,
    const unsigned short* ap0, const unsigned short* ap1,
    const unsigned short* p0, const unsigned short* p1,
    const unsigned short* p2, const unsigned short* p3, int k0)
{
    a[0] = ldfrag(ap0, k0); a[1] = ldfrag(ap1, k0);
    b[0] = ldfrag(p0, k0);  b[1] = ldfrag(p1, k0);
    b[2] = ldfrag(p2, k0);  b[3] = ldfrag(p3, k0);
}

// 8 WMMAs: 2 A-frags x 4 B-frags into 8 accumulators.
__device__ __forceinline__ void step8(v8f* acc, const v16bf* a, const v16bf* b) {
#pragma unroll
    for (int i = 0; i < 2; ++i)
#pragma unroll
        for (int j = 0; j < 4; ++j)
            acc[i * 4 + j] = wmma_bf16(a[i], b[j], acc[i * 4 + j]);
}

// ---------------- fp32 -> bf16 conversion ----------------
__global__ void cvt_bf16_kernel(const float4* __restrict__ src, uint2* __restrict__ dst, int n4) {
    int i = blockIdx.x * blockDim.x + threadIdx.x;
    int stride = gridDim.x * blockDim.x;
    for (; i < n4; i += stride) {
        float4 f = src[i];
        uint2 o;
        o.x = (unsigned)f2bf(f.x) | ((unsigned)f2bf(f.y) << 16);
        o.y = (unsigned)f2bf(f.z) | ((unsigned)f2bf(f.w) << 16);
        dst[i] = o;
    }
}

// ---------------- router: logits, top-2, softmax ----------------
__global__ __launch_bounds__(256) void router_kernel(
    const float* __restrict__ x, const float* __restrict__ Wr,
    float* __restrict__ logits_out, int* __restrict__ top_idx, float* __restrict__ top_w)
{
    int wave = threadIdx.x >> 5, lane = threadIdx.x & 31;
    int t = blockIdx.x * 8 + wave;
    int e = lane & 7, c = lane >> 3;         // 8 experts x 4 chunks of 512
    const float4* xr = (const float4*)(x + (size_t)t * HIDDEN);
    const float4* wr = (const float4*)(Wr + (size_t)e * HIDDEN);
    float p = 0.f;
    int i0 = c * 128, i1 = i0 + 128;
    for (int i = i0; i < i1; ++i) {
        float4 a = xr[i], b = wr[i];
        p += a.x * b.x + a.y * b.y + a.z * b.z + a.w * b.w;
    }
    p += __shfl_xor(p, 8, 32);
    p += __shfl_xor(p, 16, 32);
    if (lane < 8) logits_out[(size_t)t * NEXP + e] = p;

    float v[8];
#pragma unroll
    for (int i = 0; i < 8; ++i) v[i] = __shfl(p, i, 32);
    if (lane == 0) {
        int b0 = 0;
#pragma unroll
        for (int i = 1; i < 8; ++i) if (v[i] > v[b0]) b0 = i;
        int b1 = -1;
#pragma unroll
        for (int i = 0; i < 8; ++i) {
            if (i == b0) continue;
            if (b1 < 0 || v[i] > v[b1]) b1 = i;
        }
        float e1 = __expf(v[b1] - v[b0]);
        float inv = 1.f / (1.f + e1);
        top_idx[2 * t] = b0; top_idx[2 * t + 1] = b1;
        top_w[2 * t] = inv;  top_w[2 * t + 1] = e1 * inv;
    }
}

// ---------------- deterministic per-expert compaction ----------------
__global__ __launch_bounds__(256) void build_lists_kernel(
    const int* __restrict__ top_idx, const float* __restrict__ top_w,
    int* __restrict__ list, float* __restrict__ wlist, int* __restrict__ counts)
{
    int e = blockIdx.x, tid = threadIdx.x;
    __shared__ int s[256];
    __shared__ int sbase;
    if (tid == 0) sbase = 0;
    __syncthreads();
    for (int start = 0; start < TOKENS; start += 256) {
        int t = start + tid;
        int i0 = top_idx[2 * t], i1 = top_idx[2 * t + 1];
        int slot = (i0 == e) ? 0 : ((i1 == e) ? 1 : -1);
        int pred = (slot >= 0) ? 1 : 0;
        s[tid] = pred; __syncthreads();
        for (int off = 1; off < 256; off <<= 1) {
            int v = (tid >= off) ? s[tid - off] : 0;
            __syncthreads();
            s[tid] += v;
            __syncthreads();
        }
        if (pred) {
            int pos = sbase + s[tid] - 1;
            list[e * MAXM + pos] = 2 * t + slot;
            wlist[e * MAXM + pos] = top_w[2 * t + slot];
        }
        __syncthreads();
        if (tid == 0) sbase += s[255];
        __syncthreads();
    }
    if (tid == 0) counts[e] = sbase;
}

// ---------------- GEMM1: g = X@Wg^T, u = X@Wu^T, h = silu(g)*u ----------------
// block = 128 threads (4 waves); block tile 32 rows x 128 f-cols (of both g,u)
// per wave: 2 A-frags x 4 B-frags (g0,g1,u0,u1) = 8 accumulators.
// K-loop unrolled by 2 with ping-pong fragment sets (no register copies).
__global__ __launch_bounds__(128) void gemm1_kernel(
    const unsigned short* __restrict__ xbf,
    const unsigned short* __restrict__ wg,
    const unsigned short* __restrict__ wu,
    const int* __restrict__ list, const int* __restrict__ counts,
    unsigned short* __restrict__ hbuf)
{
    int e = blockIdx.x >> 7;
    int mt = blockIdx.x & 127;
    int cnt = counts[e];
    if (mt * 32 >= cnt) return;
    int hoff = 0;
#pragma unroll
    for (int i = 0; i < NEXP; ++i) hoff += (i < e) ? counts[i] : 0;

    int lane = threadIdx.x & 31, wave = threadIdx.x >> 5;
    int r = lane & 15, h = lane >> 4;
    int m0 = mt * 32;
    int rl0 = m0 + r;      if (rl0 >= cnt) rl0 = cnt - 1;
    int rl1 = m0 + 16 + r; if (rl1 >= cnt) rl1 = cnt - 1;
    int tok0 = list[e * MAXM + rl0] >> 1;
    int tok1 = list[e * MAXM + rl1] >> 1;
    const unsigned short* ap0 = xbf + (size_t)tok0 * HIDDEN + h * 8;
    const unsigned short* ap1 = xbf + (size_t)tok1 * HIDDEN + h * 8;

    int fbase = blockIdx.y * 128 + wave * 32;
    size_t wbase = (size_t)e * INTER * HIDDEN;
    const unsigned short* bg0 = wg + wbase + (size_t)(fbase + r) * HIDDEN + h * 8;
    const unsigned short* bg1 = wg + wbase + (size_t)(fbase + 16 + r) * HIDDEN + h * 8;
    const unsigned short* bu0 = wu + wbase + (size_t)(fbase + r) * HIDDEN + h * 8;
    const unsigned short* bu1 = wu + wbase + (size_t)(fbase + 16 + r) * HIDDEN + h * 8;

    v8f acc[8];
#pragma unroll
    for (int i = 0; i < 8; ++i) acc[i] = (v8f){};

    v16bf aA[2], bA[4], aB[2], bB[4];
    ldset6(aA, bA, ap0, ap1, bg0, bg1, bu0, bu1, 0);
    int k0 = 0;
    for (; k0 < HIDDEN - 64; k0 += 64) {
        ldset6(aB, bB, ap0, ap1, bg0, bg1, bu0, bu1, k0 + 32);
        step8(acc, aA, bA);
        ldset6(aA, bA, ap0, ap1, bg0, bg1, bu0, bu1, k0 + 64);
        step8(acc, aB, bB);
    }
    ldset6(aB, bB, ap0, ap1, bg0, bg1, bu0, bu1, k0 + 32);
    step8(acc, aA, bA);
    step8(acc, aB, bB);

#pragma unroll
    for (int a = 0; a < 2; ++a) {
#pragma unroll
        for (int j = 0; j < 8; ++j) {
            int m = a * 16 + j + h * 8;
            int local = m0 + m;
            if (local < cnt) {
                float g0 = acc[a * 4 + 0][j], g1 = acc[a * 4 + 1][j];
                float u0 = acc[a * 4 + 2][j], u1 = acc[a * 4 + 3][j];
                float h0 = g0 / (1.f + __expf(-g0)) * u0;
                float h1 = g1 / (1.f + __expf(-g1)) * u1;
                size_t row = (size_t)(hoff + local) * INTER;
                hbuf[row + fbase + r]      = f2bf(h0);
                hbuf[row + fbase + 16 + r] = f2bf(h1);
            }
        }
    }
}

// ---------------- GEMM2: y = h @ Wd^T, scaled scatter-add ----------------
// block = 128 threads (4 waves); block tile 32 rows x 256 d-cols
// per wave: 2 A-frags x 4 B-frags = 8 accumulators, ping-pong pipelined.
__global__ __launch_bounds__(128) void gemm2_kernel(
    const unsigned short* __restrict__ hbuf,
    const unsigned short* __restrict__ wd,
    const int* __restrict__ list, const float* __restrict__ wlist,
    const int* __restrict__ counts, float* __restrict__ out)
{
    int e = blockIdx.x >> 7;
    int mt = blockIdx.x & 127;
    int cnt = counts[e];
    if (mt * 32 >= cnt) return;
    int hoff = 0;
#pragma unroll
    for (int i = 0; i < NEXP; ++i) hoff += (i < e) ? counts[i] : 0;

    int lane = threadIdx.x & 31, wave = threadIdx.x >> 5;
    int r = lane & 15, h = lane >> 4;
    int m0 = mt * 32;
    int rl0 = m0 + r;      if (rl0 >= cnt) rl0 = cnt - 1;
    int rl1 = m0 + 16 + r; if (rl1 >= cnt) rl1 = cnt - 1;
    const unsigned short* ap0 = hbuf + (size_t)(hoff + rl0) * INTER + h * 8;
    const unsigned short* ap1 = hbuf + (size_t)(hoff + rl1) * INTER + h * 8;

    int dbase = blockIdx.y * 256 + wave * 64;
    size_t wbase = (size_t)e * HIDDEN * INTER;
    const unsigned short* b0 = wd + wbase + (size_t)(dbase + r) * INTER + h * 8;
    const unsigned short* b1 = wd + wbase + (size_t)(dbase + 16 + r) * INTER + h * 8;
    const unsigned short* b2 = wd + wbase + (size_t)(dbase + 32 + r) * INTER + h * 8;
    const unsigned short* b3 = wd + wbase + (size_t)(dbase + 48 + r) * INTER + h * 8;

    v8f acc[8];
#pragma unroll
    for (int i = 0; i < 8; ++i) acc[i] = (v8f){};

    v16bf aA[2], bA[4], aB[2], bB[4];
    ldset6(aA, bA, ap0, ap1, b0, b1, b2, b3, 0);
    int k0 = 0;
    for (; k0 < INTER - 64; k0 += 64) {
        ldset6(aB, bB, ap0, ap1, b0, b1, b2, b3, k0 + 32);
        step8(acc, aA, bA);
        ldset6(aA, bA, ap0, ap1, b0, b1, b2, b3, k0 + 64);
        step8(acc, aB, bB);
    }
    ldset6(aB, bB, ap0, ap1, b0, b1, b2, b3, k0 + 32);
    step8(acc, aA, bA);
    step8(acc, aB, bB);

#pragma unroll
    for (int a = 0; a < 2; ++a) {
#pragma unroll
        for (int j = 0; j < 8; ++j) {
            int m = a * 16 + j + h * 8;
            int local = m0 + m;
            if (local < cnt) {
                int ts = list[e * MAXM + local];
                int token = ts >> 1;
                float w = wlist[e * MAXM + local];
                float* orow = out + (size_t)token * HIDDEN + dbase + r;
                atomicAdd(orow,      acc[a * 4 + 0][j] * w);
                atomicAdd(orow + 16, acc[a * 4 + 1][j] * w);
                atomicAdd(orow + 32, acc[a * 4 + 2][j] * w);
                atomicAdd(orow + 48, acc[a * 4 + 3][j] * w);
            }
        }
    }
}

extern "C" void kernel_launch(void* const* d_in, const int* in_sizes, int n_in,
                              void* d_out, int out_size, void* d_ws, size_t ws_size,
                              hipStream_t stream) {
    const float* x  = (const float*)d_in[0];   // [4096, 2048]
    const float* Wr = (const float*)d_in[1];   // [8, 2048]
    const float* Wg = (const float*)d_in[2];   // [8, 1024, 2048]
    const float* Wu = (const float*)d_in[3];   // [8, 1024, 2048]
    const float* Wd = (const float*)d_in[4];   // [8, 2048, 1024]
    float* out = (float*)d_out;                // [4096*2048] out ++ [4096*8] logits

    char* ws = (char*)d_ws;
    size_t off = 0;
    unsigned short* xbf = (unsigned short*)(ws + off); off += (size_t)TOKENS * HIDDEN * 2;
    unsigned short* wgb = (unsigned short*)(ws + off); off += (size_t)NEXP * INTER * HIDDEN * 2;
    unsigned short* wub = (unsigned short*)(ws + off); off += (size_t)NEXP * INTER * HIDDEN * 2;
    unsigned short* wdb = (unsigned short*)(ws + off); off += (size_t)NEXP * HIDDEN * INTER * 2;
    unsigned short* hbf = (unsigned short*)(ws + off); off += (size_t)TOKENS * 2 * INTER * 2;
    int*   top_idx = (int*)(ws + off);   off += (size_t)TOKENS * 2 * 4;
    float* top_w   = (float*)(ws + off); off += (size_t)TOKENS * 2 * 4;
    int*   list    = (int*)(ws + off);   off += (size_t)NEXP * MAXM * 4;
    float* wlist   = (float*)(ws + off); off += (size_t)NEXP * MAXM * 4;
    int*   counts  = (int*)(ws + off);   off += 32;

    // zero output (final combine accumulates exactly 2 atomic adds per element)
    hipMemsetAsync(d_out, 0, (size_t)out_size * sizeof(float), stream);

    // fp32 -> bf16 conversions
    cvt_bf16_kernel<<<4096, 256, 0, stream>>>((const float4*)x,  (uint2*)xbf, TOKENS * HIDDEN / 4);
    cvt_bf16_kernel<<<4096, 256, 0, stream>>>((const float4*)Wg, (uint2*)wgb, NEXP * INTER * HIDDEN / 4);
    cvt_bf16_kernel<<<4096, 256, 0, stream>>>((const float4*)Wu, (uint2*)wub, NEXP * INTER * HIDDEN / 4);
    cvt_bf16_kernel<<<4096, 256, 0, stream>>>((const float4*)Wd, (uint2*)wdb, NEXP * HIDDEN * INTER / 4);

    // router logits + top-2 + softmax (logits live after out block)
    router_kernel<<<TOKENS / 8, 256, 0, stream>>>(x, Wr, out + (size_t)TOKENS * HIDDEN, top_idx, top_w);

    // deterministic per-expert token lists
    build_lists_kernel<<<NEXP, 256, 0, stream>>>(top_idx, top_w, list, wlist, counts);

    // expert GEMMs: 32-row M tiles halve L2 weight traffic vs 16-row tiles
    dim3 g1(NEXP * MT32, INTER / 128);
    gemm1_kernel<<<g1, 128, 0, stream>>>(xbf, wgb, wub, list, counts, hbf);
    dim3 g2(NEXP * MT32, HIDDEN / 256);
    gemm2_kernel<<<g2, 128, 0, stream>>>(hbf, wdb, list, wlist, counts, out);
}
